// MultiHeadAttention_17626545782933
// MI455X (gfx1250) — compile-verified, matrix operand order checked by reading, not run on hardware
//
#include <hip/hip_runtime.h>

// ---------------------------------------------------------------------------
// MultiHeadAttention for MI455X (gfx1250, wave32, WMMA)
// B=2, S=2048, D=1024, H=16, DH=64. bf16 WMMA w/ f32 accumulate everywhere.
// Async global->LDS double-buffered staging (ASYNCcnt-tracked).
// ---------------------------------------------------------------------------

#define BB 2
#define SS 2048
#define DD 1024
#define HH 16
#define DHC 64
#define MROWS ((BB) * (SS))   // 4096

typedef __bf16 bf16_t;
typedef __attribute__((ext_vector_type(16))) __bf16 v16bf;
typedef __attribute__((ext_vector_type(8)))  float  v8f;
typedef int i32x4 __attribute__((vector_size(16)));

union FragBF { v16bf v; uint4 q[2]; };
union H8     { uint4 q; bf16_t h[8]; };
union H4     { uint2 u; bf16_t h[4]; };

#if defined(__HIP_DEVICE_COMPILE__)
#if __has_builtin(__builtin_amdgcn_global_load_async_to_lds_b128) && \
    __has_builtin(__builtin_amdgcn_s_wait_asynccnt)
#define USE_ASYNC_COPY 1
#endif
#endif

// Copy 16 bytes global -> LDS. Async (ASYNCcnt-tracked) when available.
__device__ __forceinline__ void cp16(const bf16_t* g, bf16_t* l) {
#if USE_ASYNC_COPY
  __builtin_amdgcn_global_load_async_to_lds_b128(
      (__attribute__((address_space(1))) i32x4*)g,
      (__attribute__((address_space(3))) i32x4*)l,
      /*offset=*/0, /*cpol=*/0);
#else
  *(uint4*)l = *(const uint4*)g;
#endif
}

template <int N>
__device__ __forceinline__ void async_wait() {
#if USE_ASYNC_COPY
  __builtin_amdgcn_s_wait_asynccnt(N);   // constant required
#endif
}

__device__ __forceinline__ v8f wmma_bf16(v16bf a, v16bf b, v8f c) {
  // D = A(16x32 bf16) * B(32x16 bf16) + C(16x16 f32)
  return __builtin_amdgcn_wmma_f32_16x16x32_bf16(
      /*neg_a=*/false, a, /*neg_b=*/false, b,
      /*c_mod=*/(short)0, c, /*reuse_a=*/false, /*reuse_b=*/false);
}

// 16-bit A/B fragment (ISA 7.12.2): lane holds row/col (lane&15); its 16
// elements are two contiguous 8-element K-chunks at K = hi*8 and hi*8+16.
__device__ __forceinline__ v16bf ld_frag(const bf16_t* p, int base8) {
  FragBF f;
  f.q[0] = *(const uint4*)(p + base8);
  f.q[1] = *(const uint4*)(p + base8 + 16);
  return f.v;
}

// ---------------------------------------------------------------------------
// fp32 -> bf16 conversion (n4 float4 groups)
// ---------------------------------------------------------------------------
__global__ __launch_bounds__(256) void cvt_f32_bf16(const float* __restrict__ in,
                                                    bf16_t* __restrict__ out, int n4) {
  int i = blockIdx.x * 256 + threadIdx.x;
  if (i < n4) {
    float4 f = ((const float4*)in)[i];
    H4 t;
    t.h[0] = (bf16_t)f.x; t.h[1] = (bf16_t)f.y;
    t.h[2] = (bf16_t)f.z; t.h[3] = (bf16_t)f.w;
    ((uint2*)out)[i] = t.u;
  }
}

// fp32 W[K][N] -> bf16 W^T[N][K]  (one-time; removes in-loop LDS transpose)
__global__ __launch_bounds__(256) void cvt_w_transpose(const float* __restrict__ in,
                                                       bf16_t* __restrict__ out) {
  int t  = blockIdx.x * 256 + threadIdx.x;   // 0 .. DD*DD/4-1
  int n  = t >> 8;                           // 0..1023
  int k4 = (t & 255) * 4;                    // 0..1020
  H4 o;
#pragma unroll
  for (int j = 0; j < 4; j++) o.h[j] = (bf16_t)in[(size_t)(k4 + j) * DD + n];
  *(uint2*)&out[(size_t)n * DD + k4] = o.u;
}

// ---------------------------------------------------------------------------
// GEMM: C[4096x1024] = A[4096x1024](bf16) * W (given as W^T[N][K] bf16).
// Block tile 128x128, 8 waves (4x2), wave tile 32x64, K-step 64,
// double-buffered async LDS staging.
// MODE 0: bf16 out, head-split [B,H,S,DH]
// MODE 1: fp32 out, row-major [M,N]
// MODE 2: bf16 out, transposed head-split [B,H,DH,S]  (vectorized store)
// ---------------------------------------------------------------------------
template <int MODE>
__global__ __launch_bounds__(256) void gemm_bf16_wmma(const bf16_t* __restrict__ A,
                                                      const bf16_t* __restrict__ Bt,
                                                      void* __restrict__ Cout) {
  const int K = DD;
  __shared__ bf16_t As[2][128 * 64];   // [row][k]   16 KB x2
  __shared__ bf16_t Bs[2][128 * 64];   // [n][k]     16 KB x2

  const int tid  = threadIdx.x;
  const int wave = tid >> 5, lane = tid & 31;
  const int lo = lane & 15, hi = lane >> 4;
  const int base8 = hi * 8;
  const int wm = wave >> 1, wn = wave & 1;
  const int rowBase = blockIdx.x * 128;
  const int colBase = blockIdx.y * 128;

  const bf16_t* Ab = A  + (size_t)rowBase * K;
  const bf16_t* Bb = Bt + (size_t)colBase * K;

  // Stage: 128x64 bf16 per matrix = 1024 b128 each; 8 async ops per thread.
  auto issue = [&](int stage) {
    const int buf = stage & 1;
    const int ks  = stage * 64;
#pragma unroll
    for (int i = 0; i < 4; i++) {
      int task = tid + 256 * i;        // 0..1023
      int row  = task >> 3;            // 0..127
      int cc   = (task & 7) * 8;       // 0..56
      cp16(Ab + (size_t)row * K + ks + cc, &As[buf][row * 64 + cc]);
      cp16(Bb + (size_t)row * K + ks + cc, &Bs[buf][row * 64 + cc]);
    }
  };

  v8f acc[2][4];
  const v8f vzero = {};
#pragma unroll
  for (int mt = 0; mt < 2; mt++)
#pragma unroll
    for (int nt = 0; nt < 4; nt++) acc[mt][nt] = vzero;

  const int NST = K / 64;   // 16
  issue(0);
  for (int s = 0; s < NST; s++) {
    if (s + 1 < NST) {
      issue(s + 1);
      async_wait<8>();                 // in-order => stage s complete
    } else {
      async_wait<0>();
    }
    __syncthreads();
    const int buf = s & 1;
#pragma unroll
    for (int c = 0; c < 2; c++) {      // two K=32 chunks
      v16bf af[2], bfr[4];
#pragma unroll
      for (int mt = 0; mt < 2; mt++)
        af[mt] = ld_frag(&As[buf][(wm * 32 + mt * 16 + lo) * 64 + c * 32], base8);
#pragma unroll
      for (int nt = 0; nt < 4; nt++)
        bfr[nt] = ld_frag(&Bs[buf][(wn * 64 + nt * 16 + lo) * 64 + c * 32], base8);
#pragma unroll
      for (int mt = 0; mt < 2; mt++)
#pragma unroll
        for (int nt = 0; nt < 4; nt++)
          acc[mt][nt] = wmma_bf16(af[mt], bfr[nt], acc[mt][nt]);
    }
    __syncthreads();                   // protect buffer reuse at issue(s+2)
  }

  // Epilogue. D layout: element r -> row (r + 8*hi), col lo of the tile.
#pragma unroll
  for (int mt = 0; mt < 2; mt++)
#pragma unroll
    for (int nt = 0; nt < 4; nt++) {
      int Cc = colBase + wn * 64 + nt * 16 + lo;
      int Rb = rowBase + wm * 32 + mt * 16 + 8 * hi;   // + r, r=0..7
      if (MODE == 2) {
        // consecutive r are consecutive s -> one b128 store of 8 bf16
        int b = Rb >> 11, s0 = Rb & (SS - 1);
        int h = Cc >> 6, dh = Cc & (DHC - 1);
        H8 o;
#pragma unroll
        for (int r = 0; r < 8; r++) o.h[r] = (bf16_t)acc[mt][nt][r];
        *(uint4*)&((bf16_t*)Cout)[(((size_t)(b * HH + h)) * DHC + dh) * SS + s0] = o.q;
      } else {
#pragma unroll
        for (int r = 0; r < 8; r++) {
          int R = Rb + r;
          float v = acc[mt][nt][r];
          if (MODE == 0) {
            int b = R >> 11, s = R & (SS - 1);
            int h = Cc >> 6, dh = Cc & (DHC - 1);
            ((bf16_t*)Cout)[(((size_t)(b * HH + h)) * SS + s) * DHC + dh] = (bf16_t)v;
          } else {
            ((float*)Cout)[(size_t)R * DD + Cc] = v;
          }
        }
      }
    }
}

// ---------------------------------------------------------------------------
// Flash attention. Q/K head-split [bh][s][dh]; V pre-transposed [bh][dh][s].
// One WG = 128 query rows (8 waves x 16), 16 key blocks of 128,
// double-buffered async staging of K and V^T tiles, online softmax in f32.
// ---------------------------------------------------------------------------
__global__ __launch_bounds__(256) void attn_flash_wmma(const bf16_t* __restrict__ Q,
                                                       const bf16_t* __restrict__ Kh,
                                                       const bf16_t* __restrict__ Vtg,
                                                       const int* __restrict__ valid_lens,
                                                       bf16_t* __restrict__ ctx) {
  __shared__ bf16_t Ks[2][128 * DHC];     // [key][dh]   16 KB x2
  __shared__ bf16_t Vt[2][DHC * 128];     // [dh][key]   16 KB x2
  __shared__ bf16_t Plds[8 * 16 * 128];   // per-wave P  32 KB

  const int bh = blockIdx.y;              // 0..31
  const int b  = bh >> 4;
  const int h  = bh & (HH - 1);
  const int tid  = threadIdx.x;
  const int wave = tid >> 5, lane = tid & 31;
  const int lo = lane & 15, hi = lane >> 4;
  const int base8 = hi * 8;
  const int qbase = blockIdx.x * 128 + wave * 16;

  const bf16_t* Qb = Q   + (size_t)bh * SS * DHC;
  const bf16_t* Kb = Kh  + (size_t)bh * SS * DHC;
  const bf16_t* Vb = Vtg + (size_t)bh * DHC * SS;   // [dh][s]
  const int vlen = valid_lens[b];

  // Stage: K tile straight copy + V^T tile straight copy; 8 async ops/thread.
  auto issue = [&](int stage) {
    const int buf = stage & 1;
    const int kbase = stage * 128;
#pragma unroll
    for (int i = 0; i < 4; i++) {
      int task = tid + 256 * i;                  // 0..1023
      {
        int key = task >> 3, cc = (task & 7) * 8;
        cp16(Kb + (size_t)(kbase + key) * DHC + cc, &Ks[buf][key * DHC + cc]);
      }
      {
        int dh = task >> 4, cc = (task & 15) * 8;
        cp16(Vb + (size_t)dh * SS + kbase + cc, &Vt[buf][dh * 128 + cc]);
      }
    }
  };

  // Q fragments: A-matrix, K-dim = dh (64 -> 2 frags).
  const bf16_t* qrow = Qb + (size_t)(qbase + lo) * DHC;
  const v16bf qf0 = ld_frag(qrow,      base8);
  const v16bf qf1 = ld_frag(qrow + 32, base8);

  float m_i[8], l_i[8];
  v8f o_acc[4];
  const v8f vzero = {};
#pragma unroll
  for (int r = 0; r < 8; r++) { m_i[r] = -3.0e38f; l_i[r] = 0.0f; }
#pragma unroll
  for (int nt = 0; nt < 4; nt++) o_acc[nt] = vzero;

  bf16_t* pw = &Plds[wave * 16 * 128];

  const int NKB = SS / 128;   // 16
  issue(0);
  for (int kb = 0; kb < NKB; ++kb) {
    const int kbase = kb * 128;
    if (kb + 1 < NKB) {
      issue(kb + 1);
      async_wait<8>();
    } else {
      async_wait<0>();
    }
    __syncthreads();
    const int buf = kb & 1;

    // Scores: 8 key tiles x (2 chained WMMAs over dh).
    v8f st[8];
#pragma unroll
    for (int kt = 0; kt < 8; kt++) {
      const bf16_t* krow = &Ks[buf][(kt * 16 + lo) * DHC];
      v8f acc = vzero;
      acc = wmma_bf16(qf0, ld_frag(krow,      base8), acc);
      acc = wmma_bf16(qf1, ld_frag(krow + 32, base8), acc);
      st[kt] = acc;
    }

    // Scale + mask (reference: scores/sqrt(DH); masked -> exactly -1e6).
#pragma unroll
    for (int kt = 0; kt < 8; kt++) {
      int key = kbase + kt * 16 + lo;
      bool ok = key < vlen;
#pragma unroll
      for (int r = 0; r < 8; r++) {
        float sv = st[kt][r] * 0.125f;   // 1/sqrt(64)
        st[kt][r] = ok ? sv : -1.0e6f;
      }
    }

    // Online softmax per query row (row r+8*hi lives in one 16-lane group).
#pragma unroll
    for (int r = 0; r < 8; r++) {
      float mx = st[0][r];
#pragma unroll
      for (int kt = 1; kt < 8; kt++) mx = fmaxf(mx, st[kt][r]);
#pragma unroll
      for (int off = 8; off >= 1; off >>= 1) mx = fmaxf(mx, __shfl_xor(mx, off, 16));
      float mnew  = fmaxf(m_i[r], mx);
      float scale = __expf(m_i[r] - mnew);
      float rs = 0.0f;
#pragma unroll
      for (int kt = 0; kt < 8; kt++) {
        float p = __expf(st[kt][r] - mnew);
        st[kt][r] = p;
        rs += p;
      }
#pragma unroll
      for (int off = 8; off >= 1; off >>= 1) rs += __shfl_xor(rs, off, 16);
      l_i[r] = l_i[r] * scale + rs;
      m_i[r] = mnew;
#pragma unroll
      for (int nt = 0; nt < 4; nt++) o_acc[nt][r] *= scale;
    }

    // Re-layout P (D-layout -> A-layout) through wave-private LDS.
#pragma unroll
    for (int kt = 0; kt < 8; kt++)
#pragma unroll
      for (int r = 0; r < 8; r++)
        pw[(r + 8 * hi) * 128 + kt * 16 + lo] = (bf16_t)st[kt][r];
    __syncthreads();   // cross-lane LDS RAW

    // P(16x128) @ V(128x64): 4 K-chunks x 4 dh-tiles = 16 WMMAs.
#pragma unroll
    for (int c = 0; c < 4; c++) {
      v16bf pf = ld_frag(pw + lo * 128 + c * 32, base8);
#pragma unroll
      for (int nt = 0; nt < 4; nt++) {
        const bf16_t* vrow = &Vt[buf][(nt * 16 + lo) * 128 + c * 32];
        o_acc[nt] = wmma_bf16(pf, ld_frag(vrow, base8), o_acc[nt]);
      }
    }
    __syncthreads();   // protect buffer reuse at issue(kb+2)
  }

  // Normalize and write context row-major [B*S, D] (undoes head split).
#pragma unroll
  for (int nt = 0; nt < 4; nt++)
#pragma unroll
    for (int r = 0; r < 8; r++) {
      int q  = qbase + r + 8 * hi;
      int dh = nt * 16 + lo;
      float v = o_acc[nt][r] / l_i[r];
      ctx[((size_t)(b * SS + q)) * DD + h * DHC + dh] = (bf16_t)v;
    }
}

// ---------------------------------------------------------------------------
// Host orchestration. Workspace: 3x8MB bf16 inputs, 4x2MB bf16 W^T,
// Qh/Kh (head-split) + Vt (transposed head-split) 3x8MB, ctx 8MB = 64 MB.
// ---------------------------------------------------------------------------
extern "C" void kernel_launch(void* const* d_in, const int* in_sizes, int n_in,
                              void* d_out, int out_size, void* d_ws, size_t ws_size,
                              hipStream_t stream) {
  (void)in_sizes; (void)n_in; (void)out_size; (void)ws_size;

  const float* q_f  = (const float*)d_in[0];
  const float* k_f  = (const float*)d_in[1];
  const float* v_f  = (const float*)d_in[2];
  const int*   vlen = (const int*)d_in[3];
  const float* wq_f = (const float*)d_in[4];
  const float* wk_f = (const float*)d_in[5];
  const float* wv_f = (const float*)d_in[6];
  const float* wo_f = (const float*)d_in[7];
  float* out = (float*)d_out;

  const size_t N_IN = (size_t)MROWS * DD;   // 4,194,304
  const size_t N_W  = (size_t)DD * DD;      // 1,048,576

  bf16_t* p = (bf16_t*)d_ws;
  bf16_t* qb  = p;  p += N_IN;
  bf16_t* kb  = p;  p += N_IN;
  bf16_t* vb  = p;  p += N_IN;
  bf16_t* wqt = p;  p += N_W;
  bf16_t* wkt = p;  p += N_W;
  bf16_t* wvt = p;  p += N_W;
  bf16_t* wot = p;  p += N_W;
  bf16_t* Qh  = p;  p += N_IN;
  bf16_t* Kh  = p;  p += N_IN;
  bf16_t* Vt  = p;  p += N_IN;
  bf16_t* ctx = p;  p += N_IN;

  // fp32 -> bf16 (activations) and fp32 -> bf16 transposed (weights)
  cvt_f32_bf16<<<dim3((unsigned)(N_IN / 1024)), 256, 0, stream>>>(q_f, qb, (int)(N_IN / 4));
  cvt_f32_bf16<<<dim3((unsigned)(N_IN / 1024)), 256, 0, stream>>>(k_f, kb, (int)(N_IN / 4));
  cvt_f32_bf16<<<dim3((unsigned)(N_IN / 1024)), 256, 0, stream>>>(v_f, vb, (int)(N_IN / 4));
  cvt_w_transpose<<<dim3((unsigned)(N_W / 1024)), 256, 0, stream>>>(wq_f, wqt);
  cvt_w_transpose<<<dim3((unsigned)(N_W / 1024)), 256, 0, stream>>>(wk_f, wkt);
  cvt_w_transpose<<<dim3((unsigned)(N_W / 1024)), 256, 0, stream>>>(wv_f, wvt);
  cvt_w_transpose<<<dim3((unsigned)(N_W / 1024)), 256, 0, stream>>>(wo_f, wot);

  // Projections
  dim3 gg(MROWS / 128, DD / 128);   // (32, 8)
  gemm_bf16_wmma<0><<<gg, 256, 0, stream>>>(qb, wqt, Qh);   // Q head-split
  gemm_bf16_wmma<0><<<gg, 256, 0, stream>>>(kb, wkt, Kh);   // K head-split
  gemm_bf16_wmma<2><<<gg, 256, 0, stream>>>(vb, wvt, Vt);   // V transposed head-split

  // Flash attention
  attn_flash_wmma<<<dim3(SS / 128, BB * HH), 256, 0, stream>>>(Qh, Kh, Vt, vlen, ctx);

  // Output projection (fp32 out)
  gemm_bf16_wmma<1><<<gg, 256, 0, stream>>>(ctx, wot, out);
}
